// CustomLinearEval_34677565948161
// MI455X (gfx1250) — compile-verified, matrix operand order checked by reading, not run on hardware
//
#include <hip/hip_runtime.h>
#include <stdint.h>
#include <stddef.h>

// ---------------- problem constants (fixed by the reference) ----------------
static constexpr int Mdim = 4 * 2048;   // B*S rows of x
static constexpr int Kdim = 4096;       // in_features
static constexpr int Ndim = 4096;       // out_features
static constexpr int GS   = 128;        // quant group size along K
static constexpr float QEPS = 1e-8f;

// ---------------- tiling ----------------
static constexpr int BM = 128;
static constexpr int BN = 256;
static constexpr int BK = 64;
static constexpr int KPAD = BK + 8;          // LDS pitch in bf16 elems = 144B -> conflict-free
static constexpr int NSTAGES = Kdim / BK;    // 64

typedef __attribute__((ext_vector_type(16))) __bf16 bf16x16;
typedef __attribute__((ext_vector_type(8)))  __bf16 bf16x8;
typedef __attribute__((ext_vector_type(4)))  __bf16 bf16x4;
typedef __attribute__((ext_vector_type(8)))  float  v8f;
typedef __attribute__((ext_vector_type(4)))  float  f32x4;
typedef __attribute__((ext_vector_type(4)))  unsigned int u32x4;
typedef __attribute__((ext_vector_type(8)))  int    i32x8;
typedef __attribute__((ext_vector_type(4)))  int    i32x4;

#if defined(__has_builtin)
#  if __has_builtin(__builtin_amdgcn_tensor_load_to_lds)
#    define USE_TDM 1
#  endif
#endif
#ifndef USE_TDM
#  define USE_TDM 0
#endif

// ============================================================================
// Prepass: w_dq[n][k] = round(w[n][k]/s)*s  in fp32 (exact rounding decisions),
// stored as bf16 in workspace, layout [N][K] (== B-matrix column-major, i.e.
// n-rows of contiguous k, which is fragment-load friendly).
// ============================================================================
__global__ __launch_bounds__(256) void dequant_w_kernel(
    const float* __restrict__ w, const float* __restrict__ ss,
    __bf16* __restrict__ wq)
{
  const int idx = blockIdx.x * 256 + threadIdx.x;   // one float4 of one n-row
  const int n   = idx / (Kdim / 4);
  const int k4  = (idx % (Kdim / 4)) * 4;
  const float s = ss[(k4 / GS) * Ndim + n] + QEPS;  // step_scales is [K/GS][N]
  const f32x4 wv = *(const f32x4*)(w + (size_t)n * Kdim + k4);
  bf16x4 o;
#pragma unroll
  for (int j = 0; j < 4; ++j) {
    const float dq = rintf(wv[j] / s) * s;          // round-half-even, like jnp.round
    o[j] = (__bf16)dq;
  }
  *(bf16x4*)(wq + (size_t)n * Kdim + k4) = o;
}

// ============================================================================
// GEMM helpers
// ============================================================================
__device__ __forceinline__ bf16x16 cat8(bf16x8 lo, bf16x8 hi) {
  bf16x16 r;
#pragma unroll
  for (int i = 0; i < 8; ++i) { r[i] = lo[i]; r[i + 8] = hi[i]; }
  return r;
}

// 128 x 64 fp32 x-tile: 256 threads * 8 float4 loads
__device__ __forceinline__ void x_issue(const float* __restrict__ x, int bm, int k0,
                                        int tid, f32x4 (&xr)[8]) {
#pragma unroll
  for (int i = 0; i < 8; ++i) {
    const int q  = tid + i * 256;   // 2048 float4 chunks
    const int m  = q >> 4;
    const int kq = (q & 15) << 2;
    xr[i] = *(const f32x4*)(x + (size_t)(bm + m) * Kdim + k0 + kq);
  }
}

__device__ __forceinline__ void x_store(__bf16* __restrict__ dst,
                                        const f32x4 (&xr)[8], int tid) {
#pragma unroll
  for (int i = 0; i < 8; ++i) {
    const int q  = tid + i * 256;
    const int m  = q >> 4;
    const int kq = (q & 15) << 2;
    bf16x4 h;
#pragma unroll
    for (int j = 0; j < 4; ++j) h[j] = (__bf16)xr[i][j];
    *(bf16x4*)(dst + m * KPAD + kq) = h;  // ds_store_b64
  }
}

#if USE_TDM
// TDM: one descriptor moves a BN x BK bf16 tile (n-rows, contiguous k) from the
// dequantized weight matrix into LDS, padding each 128B row by 16B so the LDS
// pitch is KPAD (bank-conflict-free fragment reads). One issue per stage by
// wave 0; completion tracked with TENSORcnt.
__device__ __forceinline__ void tdm_load_w(const __bf16* __restrict__ wq, int bn, int k0,
                                           __bf16* __restrict__ ldsw) {
  const uint64_t gaddr = (uint64_t)(uintptr_t)(wq + (size_t)bn * Kdim + k0);
  const uint32_t laddr = (uint32_t)(uintptr_t)ldsw; // low 32 bits of flat addr = LDS offset
  u32x4 g0;
  g0.x = 1u;                                        // count=1, user descriptor
  g0.y = laddr;                                     // lds_addr  [63:32]
  g0.z = (uint32_t)gaddr;                           // global_addr lo
  g0.w = (uint32_t)((gaddr >> 32) & 0x01FFFFFFu) | (2u << 30); // addr hi | type=2
  i32x8 g1;
  g1[0] = (int)((1u << 16)     // data_size = 2 bytes
              | (1u << 20)     // pad_enable
              | (4u << 22)     // pad_interval: 32 dwords (= one 64-elem bf16 row)
              | (3u << 25));   // pad_amount:   4 dwords (= 8 bf16) -> KPAD pitch
  g1[1] = (int)((uint32_t)Kdim << 16);                              // tensor_dim0 lo16
  g1[2] = (int)(((uint32_t)Kdim >> 16) | ((uint32_t)Ndim << 16));   // dim0 hi | dim1 lo
  g1[3] = (int)(((uint32_t)Ndim >> 16) | ((uint32_t)BK << 16));     // dim1 hi | tile_dim0
  g1[4] = BN;                                                       // tile_dim1 (tile_dim2=0)
  g1[5] = Kdim;                                                     // tensor_dim0_stride lo32
  g1[6] = 0;
  g1[7] = 0;
  const i32x4 gz4 = {0, 0, 0, 0};
  const i32x8 gz8 = {0, 0, 0, 0, 0, 0, 0, 0};
  // 6-arg toolchain variant: (g0, g1, g2, g3, g4, cpol)
  __builtin_amdgcn_tensor_load_to_lds(g0, g1, gz4, gz4, gz8, 0);
}
#else
// Fallback: plain vector-copy of the weight tile into the same padded layout.
__device__ __forceinline__ void w_copy(const __bf16* __restrict__ wq, int bn, int k0,
                                       __bf16* __restrict__ dst, int tid) {
#pragma unroll
  for (int i = 0; i < 8; ++i) {
    const int q  = tid + i * 256;   // 2048 chunks of 8 bf16 = 256 x 64
    const int n  = q >> 3;
    const int kq = (q & 7) << 3;
    *(bf16x8*)(dst + n * KPAD + kq) =
        *(const bf16x8*)(wq + (size_t)(bn + n) * Kdim + k0 + kq);
  }
}
#endif

// ============================================================================
// GEMM: out[M][N] = x[M][K] (bf16, on the fly) * wq[K][N] (bf16) + bias
// Workgroup: 256 threads (8 waves), block tile 128x256, BK=64, double-buffered.
// Wave tile 64(M) x 64(N): 4 A-frags x 4 B-frags -> 16 wmma per 32-k step,
// 16 independent accumulators (no WMMA->WMMA RAW hazards).
// ============================================================================
__global__ __launch_bounds__(256, 1) void gemm_wmma_kernel(
    const float* __restrict__ x, const __bf16* __restrict__ wq,
    const float* __restrict__ bias, float* __restrict__ out)
{
  __shared__ __bf16 ldsX[2][BM * KPAD];
  __shared__ __bf16 ldsW[2][BN * KPAD];

  const int tid  = threadIdx.x;
  const int lane = tid & 31;
  const int lr   = lane & 15;                        // row within 16
  const int lh   = lane >> 4;                        // k-half selector
  const int wid  = __builtin_amdgcn_readfirstlane(tid) >> 5;  // scalar wave id
  const int wm   = (wid & 1) * 64;                   // wave M offset in block tile
  const int wn   = (wid >> 1) * 64;                  // wave N offset in block tile

  const int bm = blockIdx.x * BM;
  const int bn = blockIdx.y * BN;

  v8f acc[4][4] = {};

  // ---- prologue: stage 0 into buffer 0 ----
  {
    f32x4 x0[8];
    x_issue(x, bm, 0, tid, x0);
    x_store(&ldsX[0][0], x0, tid);
  }
#if USE_TDM
  if (wid == 0) {
    tdm_load_w(wq, bn, 0, &ldsW[0][0]);
    __builtin_amdgcn_s_wait_tensorcnt(0);
  }
#else
  w_copy(wq, bn, 0, &ldsW[0][0], tid);
#endif
  __syncthreads();

  int buf = 0;
  f32x4 xr[8];
  for (int s = 0; s < NSTAGES; ++s) {
    const int nbuf = buf ^ 1;
    const bool has_next = (s + 1 < NSTAGES);

    if (has_next) {
      x_issue(x, bm, (s + 1) * BK, tid, xr);   // global loads overlap compute
#if USE_TDM
      if (wid == 0) tdm_load_w(wq, bn, (s + 1) * BK, &ldsW[nbuf][0]);
#endif
    }

#pragma unroll
    for (int kk = 0; kk < BK; kk += 32) {
      bf16x16 afrag[4];
      bf16x16 bfrag[4];
#pragma unroll
      for (int f = 0; f < 4; ++f) {
        const __bf16* p = &ldsX[buf][(wm + f * 16 + lr) * KPAD + kk + lh * 8];
        afrag[f] = cat8(*(const bf16x8*)p, *(const bf16x8*)(p + 16));
      }
#pragma unroll
      for (int g = 0; g < 4; ++g) {
        const __bf16* p = &ldsW[buf][(wn + g * 16 + lr) * KPAD + kk + lh * 8];
        bfrag[g] = cat8(*(const bf16x8*)p, *(const bf16x8*)(p + 16));
      }
#pragma unroll
      for (int f = 0; f < 4; ++f)
#pragma unroll
        for (int g = 0; g < 4; ++g)
          acc[f][g] = __builtin_amdgcn_wmma_f32_16x16x32_bf16(
              false, afrag[f], false, bfrag[g], (short)0, acc[f][g], false, false);
    }

    if (has_next) {
      x_store(&ldsX[nbuf][0], xr, tid);
#if USE_TDM
      if (wid == 0) __builtin_amdgcn_s_wait_tensorcnt(0);
#else
      w_copy(wq, bn, (s + 1) * BK, &ldsW[nbuf][0], tid);
#endif
    }
    __syncthreads();
    buf = nbuf;
  }

  // ---- epilogue: C/D layout -> lanes 0-15: M=r, N=lane; lanes 16-31: M=8+r ----
#pragma unroll
  for (int f = 0; f < 4; ++f) {
#pragma unroll
    for (int g = 0; g < 4; ++g) {
      const int n  = bn + wn + g * 16 + lr;
      const float bv = bias[n];
#pragma unroll
      for (int r = 0; r < 8; ++r) {
        const int m = bm + wm + f * 16 + lh * 8 + r;
        __builtin_nontemporal_store(acc[f][g][r] + bv, out + (size_t)m * Ndim + n);
      }
    }
  }
}

// ============================================================================
// Launch
// ============================================================================
extern "C" void kernel_launch(void* const* d_in, const int* in_sizes, int n_in,
                              void* d_out, int out_size, void* d_ws, size_t ws_size,
                              hipStream_t stream) {
  const float* x    = (const float*)d_in[0];   // [B,S,K] = [8192,4096] f32
  const float* w    = (const float*)d_in[1];   // [N,K] f32
  const float* bias = (const float*)d_in[2];   // [N]   f32
  const float* ss   = (const float*)d_in[3];   // [K/GS,N] f32
  float* out        = (float*)d_out;           // [8192,4096] f32
  __bf16* wq        = (__bf16*)d_ws;           // needs N*K*2 = 32 MB scratch

  dequant_w_kernel<<<(Ndim * (Kdim / 4)) / 256, 256, 0, stream>>>(w, ss, wq);

  dim3 grid(Mdim / BM, Ndim / BN);             // 64 x 16 workgroups
  gemm_wmma_kernel<<<grid, dim3(256), 0, stream>>>(x, wq, bias, out);
}
